// SingleHeadAttention_8461085573457
// MI455X (gfx1250) — matrix operand order repaired and checked by reading, not compiled, not run on hardware
//
#include <hip/hip_runtime.h>
#include <hip/hip_bf16.h>
#include <stdint.h>

// ---------------- problem constants ----------------
#define B_     32
#define T_MEL  2048
#define T_TEXT 512
#define N_MEL  80
#define N_TEXT 256
#define D_ATT  80
#define D_PAD  96     // K padded to multiple of 32 for wmma 16x16x32
#define N_OUT  256
#define M_TILE 32     // mel rows per block (2 x 16-row WMMA subtiles)

typedef __attribute__((ext_vector_type(16))) _Float16 v16h;
typedef __attribute__((ext_vector_type(8)))  float    v8f;

union FragH { v16h v; uint4 q[2]; _Float16 h[16]; };
union Half4 { _Float16 h[4]; uint2 u; };

static __device__ __forceinline__ v8f wmma_f16(v16h a, v16h b, v8f c) {
  // D = A(16x32 f16) * B(32x16 f16) + C(16x16 f32)
  return __builtin_amdgcn_wmma_f32_16x16x32_f16(
      /*neg_a=*/false, a, /*neg_b=*/false, b,
      /*c_mod=*/(short)0, c, /*reuse_a=*/false, /*reuse_b=*/false);
}

// A fragment: 16x32 f16 tile from row-major [row][k] LDS buffer (stride in halves).
static __device__ __forceinline__ v16h load_A(const _Float16* src, int stride, int row0, int k0, int lane) {
  const int row  = row0 + (lane & 15);
  const int koff = (lane >> 4) << 3;            // 0 or 8
  FragH f;
  const _Float16* p = src + row * stride + k0 + koff;
  f.q[0] = *(const uint4*)(p);                  // K = k0+koff .. +7
  f.q[1] = *(const uint4*)(p + 16);             // K = k0+16+koff .. +7
  return f.v;
}

// B fragment: 32x16 f16 tile (KxN) from [n][k] LDS buffer (k contiguous).
static __device__ __forceinline__ v16h load_B(const _Float16* src, int stride, int n0, int k0, int lane) {
  const int n    = n0 + (lane & 15);
  const int koff = (lane >> 4) << 3;            // 0 or 8
  FragH f;
  const _Float16* p = src + n * stride + k0 + koff;
  f.q[0] = *(const uint4*)(p);
  f.q[1] = *(const uint4*)(p + 16);
  return f.v;
}

// CDNA5 async memory->LDS copy, 16B per lane, tracked by ASYNCcnt (no VGPR data).
static __device__ __forceinline__ void async_load_b128(uint32_t lds_byte_off,
                                                       uint64_t gbase, uint32_t voff) {
  asm volatile("global_load_async_to_lds_b128 %0, %1, %2"
               :: "v"(lds_byte_off), "v"(voff), "s"(gbase) : "memory");
}
static __device__ __forceinline__ void wait_async0() {
  asm volatile("s_wait_asynccnt 0" ::: "memory");
}
static __device__ __forceinline__ uint32_t lds_off(const void* p) {
  // generic pointer to LDS: low 32 bits are the LDS offset (flat aperture truncation)
  return (uint32_t)(uintptr_t)p;
}

// ---------------- projection kernels ----------------
// q = (queries @ Wq + bq) * 80^-0.25  -> f16, K padded to 96 with zeros
__global__ void proj_q_kernel(const float* __restrict__ qin, const float* __restrict__ Wq,
                              const float* __restrict__ bq, _Float16* __restrict__ qbuf) {
  __shared__ float sX[32 * N_MEL];
  __shared__ float sW[N_MEL * D_ATT];
  __shared__ float sB[D_ATT];
  const int tid = threadIdx.x;
  const int row0 = blockIdx.x * 32;
  for (int i = tid; i < 32 * N_MEL; i += blockDim.x) sX[i] = qin[(size_t)row0 * N_MEL + i];
  for (int i = tid; i < N_MEL * D_ATT; i += blockDim.x) sW[i] = Wq[i];
  for (int i = tid; i < D_ATT; i += blockDim.x) sB[i] = bq[i];
  __syncthreads();
  const float invScale = 0.33437015f;           // 80^-0.25
  for (int idx = tid; idx < 32 * D_ATT; idx += blockDim.x) {
    const int r = idx / D_ATT, d = idx % D_ATT;
    float acc = sB[d];
    for (int i = 0; i < N_MEL; ++i) acc += sX[r * N_MEL + i] * sW[i * D_ATT + d];
    qbuf[(size_t)(row0 + r) * D_PAD + d] = (_Float16)(acc * invScale);
  }
  for (int idx = tid; idx < 32 * (D_PAD - D_ATT); idx += blockDim.x) {
    const int r = idx >> 4, d = D_ATT + (idx & 15);
    qbuf[(size_t)(row0 + r) * D_PAD + d] = (_Float16)0.0f;
  }
}

// k = (keys @ Wk + bk)*scale -> f16 [row][96]; v = values @ Wv + bv -> f16 TRANSPOSED [b][d][t]
__global__ void proj_kv_kernel(const float* __restrict__ keys, const float* __restrict__ values,
                               const float* __restrict__ Wk, const float* __restrict__ bk,
                               const float* __restrict__ Wv, const float* __restrict__ bv,
                               _Float16* __restrict__ kbuf, _Float16* __restrict__ vTbuf) {
  __shared__ float sK[N_TEXT];
  __shared__ float sV[N_TEXT];
  const int tid = threadIdx.x;
  const int row = blockIdx.x;                   // b*512 + t
  const int b = row / T_TEXT, t = row % T_TEXT;
  if (tid < N_TEXT) {
    sK[tid] = keys  [(size_t)row * N_TEXT + tid];
    sV[tid] = values[(size_t)row * N_TEXT + tid];
  }
  __syncthreads();
  const float invScale = 0.33437015f;
  if (tid < D_ATT) {
    const int d = tid;
    float acc = bk[d];
    for (int i = 0; i < N_TEXT; ++i) acc += sK[i] * Wk[i * D_ATT + d];
    kbuf[(size_t)row * D_PAD + d] = (_Float16)(acc * invScale);
  } else if (tid < 2 * D_ATT) {
    const int d = tid - D_ATT;
    float acc = bv[d];
    for (int i = 0; i < N_TEXT; ++i) acc += sV[i] * Wv[i * D_ATT + d];
    vTbuf[((size_t)b * D_ATT + d) * T_TEXT + t] = (_Float16)acc;
  } else if (tid < 2 * D_ATT + (D_PAD - D_ATT)) {
    kbuf[(size_t)row * D_PAD + D_ATT + (tid - 2 * D_ATT)] = (_Float16)0.0f;
  }
}

// WoT[n][k] = Wo[k][n] (f16, k padded 80->96 with zeros)
__global__ void prep_wo_kernel(const float* __restrict__ Wo, _Float16* __restrict__ WoT) {
  for (int idx = threadIdx.x; idx < N_OUT * D_PAD; idx += blockDim.x) {
    const int n = idx / D_PAD, k = idx % D_PAD;
    WoT[idx] = (k < D_ATT) ? (_Float16)Wo[k * N_OUT + n] : (_Float16)0.0f;
  }
}

// ---------------- fused attention ----------------
// One block = (batch, 32-row mel tile). 256 threads = 8 wave32s.
// Dynamic LDS (~252 KB, CDNA5 allows 320 KB per workgroup).
// sS is placed at offset 0 so all score-tile ds_store offsets (rows * 2KB + col)
// fit the 16-bit DS immediate-offset field -> no per-store address VALU.
//   sS  : scores 32x512 f32 (64 KB)           @ 0
//   sKV : k panel 512x96 f16 (96 KB), reused for v^T 80x512 f16 (80 KB)
//   sWo : Wo^T 256x96 f16 (48 KB)
//   sP  : probs 32x512 f16 (32 KB)
//   sQ  : q tile 32x96 f16 (6 KB)
//   sC  : context tile 32x96 f16 (6 KB)
#define SM_S  0
#define SM_KV (SM_S + 65536)
#define SM_WO (SM_KV + 98304)
#define SM_P  (SM_WO + 49152)
#define SM_Q  (SM_P + 32768)
#define SM_C  (SM_Q + 6144)
#define SM_BYTES (SM_C + 6144)

__global__ __launch_bounds__(256) void attention_kernel(
    const _Float16* __restrict__ qbuf, const _Float16* __restrict__ kbuf,
    const _Float16* __restrict__ vTbuf, const _Float16* __restrict__ WoT,
    const float* __restrict__ prior, const unsigned char* __restrict__ amask,
    const float* __restrict__ bo,
    float* __restrict__ out_states, float* __restrict__ attn1, float* __restrict__ attn2) {
  extern __shared__ __align__(16) char smem[];
  float*    sS  = (float*)   (smem + SM_S);
  _Float16* sKV = (_Float16*)(smem + SM_KV);
  _Float16* sWo = (_Float16*)(smem + SM_WO);
  _Float16* sP  = (_Float16*)(smem + SM_P);
  _Float16* sQ  = (_Float16*)(smem + SM_Q);
  _Float16* sC  = (_Float16*)(smem + SM_C);

  const int tid  = threadIdx.x;
  const int lane = tid & 31;
  const int wave = tid >> 5;
  const int b     = blockIdx.y;
  const int mbase = blockIdx.x * M_TILE;

  // ---- phase 1: async-stage q tile + full k panel + WoT into LDS ----
  {
    const uint64_t gq = (uint64_t)(uintptr_t)(qbuf + (size_t)(b * T_MEL + mbase) * D_PAD);
    const uint32_t dq = lds_off(sQ);
    for (int i = tid; i < (M_TILE * D_PAD) / 8; i += 256)          // 384 x 16B
      async_load_b128(dq + i * 16, gq, (uint32_t)(i * 16));
    const uint64_t gk = (uint64_t)(uintptr_t)(kbuf + (size_t)b * T_TEXT * D_PAD);
    const uint32_t dk = lds_off(sKV);
    for (int i = tid; i < (T_TEXT * D_PAD) / 8; i += 256)          // 6144 x 16B
      async_load_b128(dk + i * 16, gk, (uint32_t)(i * 16));
    const uint64_t gw = (uint64_t)(uintptr_t)WoT;
    const uint32_t dw = lds_off(sWo);
    for (int i = tid; i < (N_OUT * D_PAD) / 8; i += 256)           // 3072 x 16B
      async_load_b128(dw + i * 16, gw, (uint32_t)(i * 16));
    // prefetch the prior panel for this block while WMMAs run (global_prefetch_b8)
    const float* pbase = prior + (size_t)(b * T_MEL + mbase) * T_TEXT;
    __builtin_prefetch(pbase + tid * 64, 0, 0);
    __builtin_prefetch(pbase + tid * 64 + 32, 0, 0);
    wait_async0();
  }
  __syncthreads();

  // ---- phase 2: scores = q @ k^T (32 x 512), K = 96, WMMA ----
  // Both 16-row m-subtiles per N-tile: each B fragment feeds 2 WMMAs.
  {
    v16h aq0_0 = load_A(sQ, D_PAD, 0, 0, lane);
    v16h aq0_1 = load_A(sQ, D_PAD, 0, 32, lane);
    v16h aq0_2 = load_A(sQ, D_PAD, 0, 64, lane);
    v16h aq1_0 = load_A(sQ, D_PAD, 16, 0, lane);
    v16h aq1_1 = load_A(sQ, D_PAD, 16, 32, lane);
    v16h aq1_2 = load_A(sQ, D_PAD, 16, 64, lane);
    for (int nt = wave; nt < T_TEXT / 16; nt += 8) {               // 4 N-tiles per wave
      v16h bm0 = load_B(sKV, D_PAD, nt * 16, 0, lane);
      v16h bm1 = load_B(sKV, D_PAD, nt * 16, 32, lane);
      v16h bm2 = load_B(sKV, D_PAD, nt * 16, 64, lane);
      v8f acc0 = {0.f, 0.f, 0.f, 0.f, 0.f, 0.f, 0.f, 0.f};
      v8f acc1 = {0.f, 0.f, 0.f, 0.f, 0.f, 0.f, 0.f, 0.f};
      acc0 = wmma_f16(aq0_0, bm0, acc0);
      acc1 = wmma_f16(aq1_0, bm0, acc1);
      acc0 = wmma_f16(aq0_1, bm1, acc0);
      acc1 = wmma_f16(aq1_1, bm1, acc1);
      acc0 = wmma_f16(aq0_2, bm2, acc0);
      acc1 = wmma_f16(aq1_2, bm2, acc1);
      const int n  = nt * 16 + (lane & 15);
      const int mb = (lane >> 4) << 3;
#pragma unroll
      for (int r = 0; r < 8; ++r) {
        sS[(mb + r) * T_TEXT + n]      = acc0[r];
        sS[(16 + mb + r) * T_TEXT + n] = acc1[r];
      }
    }
  }
  __syncthreads();

  // ---- phase 3a: issue async v^T fetch into sKV NOW (k panel dead after the
  //      barrier above); completion hidden under the softmax, waited in 3c ----
  {
    const uint64_t gv = (uint64_t)(uintptr_t)(vTbuf + (size_t)b * D_ATT * T_TEXT);
    const uint32_t dv = lds_off(sKV);
    for (int i = tid; i < (D_ATT * T_TEXT) / 8; i += 256)          // 5120 x 16B
      async_load_b128(dv + i * 16, gv, (uint32_t)(i * 16));
  }

  // ---- phase 3b: soft_attn = normalize((prior+1e-8) * exp(s - smax)), masked.
  // softmax(log_softmax(s)+log(p)) == softmax(s+log p); using smax (masked row
  // max of s alone) as stabilizer keeps exp() <= 1 — no logf needed at all.
  {
    const int m  = tid >> 3;                                       // 8 threads per row
    const int l8 = tid & 7;                                        // 4-col chunks
    float* srow = sS + m * T_TEXT;
    const float4* prow4 = (const float4*)(prior + ((size_t)(b * T_MEL + mbase + m)) * T_TEXT);
    const uchar4* mrow4 = (const uchar4*)(amask + (size_t)b * T_TEXT);
    float xmax = -INFINITY;
#pragma unroll 4
    for (int i = 0; i < T_TEXT / 32; ++i) {
      const int c4 = i * 8 + l8;
      const float4 s4 = ((const float4*)srow)[c4];
      const uchar4 mk = mrow4[c4];
      xmax = fmaxf(xmax, fmaxf(fmaxf(mk.x ? s4.x : -INFINITY, mk.y ? s4.y : -INFINITY),
                               fmaxf(mk.z ? s4.z : -INFINITY, mk.w ? s4.w : -INFINITY)));
    }
    for (int off = 4; off >= 1; off >>= 1) xmax = fmaxf(xmax, __shfl_xor(xmax, off, 8));
    float esum = 0.f;
#pragma unroll 4
    for (int i = 0; i < T_TEXT / 32; ++i) {
      const int c4 = i * 8 + l8;
      const float4 s4 = ((const float4*)srow)[c4];
      const float4 p4 = prow4[c4];
      const uchar4 mk = mrow4[c4];
      float4 e4;
      e4.x = mk.x ? (p4.x + 1e-8f) * __expf(s4.x - xmax) : 0.f;
      e4.y = mk.y ? (p4.y + 1e-8f) * __expf(s4.y - xmax) : 0.f;
      e4.z = mk.z ? (p4.z + 1e-8f) * __expf(s4.z - xmax) : 0.f;
      e4.w = mk.w ? (p4.w + 1e-8f) * __expf(s4.w - xmax) : 0.f;
      ((float4*)srow)[c4] = e4;
      esum += (e4.x + e4.y) + (e4.z + e4.w);
    }
    for (int off = 4; off >= 1; off >>= 1) esum += __shfl_xor(esum, off, 8);
    const float inv = (esum > 0.f) ? 1.f / esum : 0.f;
    float4* a1row = (float4*)(attn1 + ((size_t)(b * T_MEL + mbase + m)) * T_TEXT);
    float4* a2row = (float4*)(attn2 + ((size_t)(b * T_MEL + mbase + m)) * T_TEXT);
    uint2*  hrow  = (uint2*)(sP + m * T_TEXT);
#pragma unroll 4
    for (int i = 0; i < T_TEXT / 32; ++i) {
      const int c4 = i * 8 + l8;
      float4 e4 = ((const float4*)srow)[c4];
      e4.x *= inv; e4.y *= inv; e4.z *= inv; e4.w *= inv;
      a1row[c4] = e4;
      a2row[c4] = e4;
      Half4 h;
      h.h[0] = (_Float16)e4.x; h.h[1] = (_Float16)e4.y;
      h.h[2] = (_Float16)e4.z; h.h[3] = (_Float16)e4.w;
      hrow[c4] = h.u;
    }
  }

  // ---- phase 3c: v^T async copy must have landed before WMMA reads it ----
  wait_async0();
  __syncthreads();

  // ---- phase 5: context = P @ v (32 x 80), K = 512, WMMA ----
  // 5 waves; each keeps two accumulators so one B fragment feeds 2 WMMAs.
  if (wave < D_ATT / 16) {
    const int nbase = wave * 16;
    v8f acc0 = {0.f, 0.f, 0.f, 0.f, 0.f, 0.f, 0.f, 0.f};
    v8f acc1 = {0.f, 0.f, 0.f, 0.f, 0.f, 0.f, 0.f, 0.f};
#pragma unroll 4
    for (int kk = 0; kk < T_TEXT; kk += 32) {
      v16h bm = load_B(sKV, T_TEXT, nbase, kk, lane);
      acc0 = wmma_f16(load_A(sP, T_TEXT, 0, kk, lane), bm, acc0);
      acc1 = wmma_f16(load_A(sP, T_TEXT, 16, kk, lane), bm, acc1);
    }
    const int n  = nbase + (lane & 15);
    const int mb = (lane >> 4) << 3;
#pragma unroll
    for (int r = 0; r < 8; ++r) {
      sC[(mb + r) * D_PAD + n]      = (_Float16)acc0[r];
      sC[(16 + mb + r) * D_PAD + n] = (_Float16)acc1[r];
    }
  }
  // zero-pad context columns 80..95 (32x16 pad cells, 2 per thread)
  {
    const int r0 = tid >> 4, c0 = tid & 15;
    sC[r0 * D_PAD + D_ATT + c0] = (_Float16)0.0f;
    sC[(r0 + 16) * D_PAD + D_ATT + c0] = (_Float16)0.0f;
  }
  __syncthreads();

  // ---- phase 6: out = context @ Wo + bo (32 x 256), K = 96, WMMA ----
  {
    v16h ac0_0 = load_A(sC, D_PAD, 0, 0, lane);
    v16h ac0_1 = load_A(sC, D_PAD, 0, 32, lane);
    v16h ac0_2 = load_A(sC, D_PAD, 0, 64, lane);
    v16h ac1_0 = load_A(sC, D_PAD, 16, 0, lane);
    v16h ac1_1 = load_A(sC, D_PAD, 16, 32, lane);
    v16h ac1_2 = load_A(sC, D_PAD, 16, 64, lane);
    for (int nt = wave; nt < N_OUT / 16; nt += 8) {                // 2 N-tiles per wave
      const int nbase = nt * 16;
      v16h bm0 = load_B(sWo, D_PAD, nbase, 0, lane);
      v16h bm1 = load_B(sWo, D_PAD, nbase, 32, lane);
      v16h bm2 = load_B(sWo, D_PAD, nbase, 64, lane);
      v8f acc0 = {0.f, 0.f, 0.f, 0.f, 0.f, 0.f, 0.f, 0.f};
      v8f acc1 = {0.f, 0.f, 0.f, 0.f, 0.f, 0.f, 0.f, 0.f};
      acc0 = wmma_f16(ac0_0, bm0, acc0);
      acc1 = wmma_f16(ac1_0, bm0, acc1);
      acc0 = wmma_f16(ac0_1, bm1, acc0);
      acc1 = wmma_f16(ac1_1, bm1, acc1);
      acc0 = wmma_f16(ac0_2, bm2, acc0);
      acc1 = wmma_f16(ac1_2, bm2, acc1);
      const int n    = nbase + (lane & 15);
      const float bb = bo[n];
      const int mb   = (lane >> 4) << 3;
#pragma unroll
      for (int r = 0; r < 8; ++r) {
        out_states[((size_t)(b * T_MEL + mbase + mb + r)) * N_OUT + n]      = acc0[r] + bb;
        out_states[((size_t)(b * T_MEL + mbase + 16 + mb + r)) * N_OUT + n] = acc1[r] + bb;
      }
    }
  }
}

// ---------------- launch ----------------
extern "C" void kernel_launch(void* const* d_in, const int* in_sizes, int n_in,
                              void* d_out, int out_size, void* d_ws, size_t ws_size,
                              hipStream_t stream) {
  (void)in_sizes; (void)n_in; (void)out_size; (void)ws_size;
  const float* queries = (const float*)d_in[0];
  const float* keys    = (const float*)d_in[1];
  const float* values  = (const float*)d_in[2];
  const unsigned char* amask = (const unsigned char*)d_in[3]; // jnp bool = 1 byte
  const float* prior   = (const float*)d_in[4];
  const float* Wq = (const float*)d_in[5];
  const float* bq = (const float*)d_in[6];
  const float* Wk = (const float*)d_in[7];
  const float* bk = (const float*)d_in[8];
  const float* Wv = (const float*)d_in[9];
  const float* bv = (const float*)d_in[10];
  const float* Wo = (const float*)d_in[11];
  const float* bo = (const float*)d_in[12];

  // workspace: q f16 [65536][96], k f16 [16384][96], v^T f16 [32][80][512], Wo^T f16 [256][96]
  _Float16* qbuf = (_Float16*)d_ws;
  _Float16* kbuf = qbuf + (size_t)B_ * T_MEL * D_PAD;
  _Float16* vT   = kbuf + (size_t)B_ * T_TEXT * D_PAD;
  _Float16* WoT  = vT   + (size_t)B_ * D_ATT * T_TEXT;   // total ~18.4 MB

  float* out   = (float*)d_out;
  float* attn1 = out   + (size_t)B_ * T_MEL * N_OUT;
  float* attn2 = attn1 + (size_t)B_ * T_MEL * T_TEXT;

  proj_q_kernel <<<(B_ * T_MEL) / 32, 256, 0, stream>>>(queries, Wq, bq, qbuf);
  proj_kv_kernel<<<B_ * T_TEXT, 256, 0, stream>>>(keys, values, Wk, bk, Wv, bv, kbuf, vT);
  prep_wo_kernel<<<1, 256, 0, stream>>>(Wo, WoT);
  attention_kernel<<<dim3(T_MEL / M_TILE, B_), 256, SM_BYTES, stream>>>(
      qbuf, kbuf, vT, WoT, prior, amask, bo, out, attn1, attn2);
}
